// DelocalizedEmbedSparse_35338990911511
// MI455X (gfx1250) — matrix operand order repaired and checked by reading, not compiled, not run on hardware
//
#include <hip/hip_runtime.h>
#include <hip/hip_bf16.h>

// Problem sizes (match reference)
#define NATOMS 131072
#define NGRAPH 2048
#define NFEAT  256

typedef __attribute__((ext_vector_type(16))) _Float16 v16h;
typedef __attribute__((ext_vector_type(8)))  _Float16 v8h;
typedef __attribute__((ext_vector_type(8)))  float    v8f;

__device__ __forceinline__ float silu_f(float x) {
    return x / (1.0f + __expf(-x));
}

// ---------------------------------------------------------------------------
// prep_c: c[j] = W_q @ k_table[j]  (two F-vectors) -- removes the first GEMM
// ---------------------------------------------------------------------------
__global__ void prep_c_kernel(const float* __restrict__ Wq,
                              const float* __restrict__ ktab,
                              float* __restrict__ c) {
    int f = threadIdx.x;  // 256 threads
    float s0 = 0.f, s1 = 0.f;
    for (int j = 0; j < NFEAT; ++j) {
        float w = Wq[f * NFEAT + j];
        s0 += w * ktab[j];
        s1 += w * ktab[NFEAT + j];
    }
    c[f] = s0;
    c[NFEAT + f] = s1;
}

// ---------------------------------------------------------------------------
// prep_g: psi_idx[g] = (int)floor(psi/inf)  (0 for finite psi >= 0)
// ---------------------------------------------------------------------------
__global__ void prep_g_kernel(const float* __restrict__ psi,
                              int* __restrict__ idx_g) {
    int g = blockIdx.x * blockDim.x + threadIdx.x;
    if (g < NGRAPH) {
        float q = psi[g] / __builtin_huge_valf();
        idx_g[g] = (int)floorf(q);
    }
}

// ---------------------------------------------------------------------------
// conv_w: convert residual-MLP weights f32 -> f16 once (halves L2 traffic)
// ---------------------------------------------------------------------------
__global__ void conv_w_kernel(const float* __restrict__ W1,
                              const float* __restrict__ W2,
                              _Float16* __restrict__ W1h,
                              _Float16* __restrict__ W2h) {
    int i = blockIdx.x * blockDim.x + threadIdx.x;  // NFEAT*NFEAT threads
    W1h[i] = (_Float16)W1[i];
    W2h[i] = (_Float16)W2[i];
}

// ---------------------------------------------------------------------------
// score_y: y[i] = softplus( (e_Z[i] . c[idx[seg[i]]]) / sqrt(F) )
// one wave (32 lanes) per atom row; float4 loads; xor-shuffle reduction
// ---------------------------------------------------------------------------
__global__ __launch_bounds__(256)
void score_y_kernel(const float* __restrict__ eZ,
                    const int* __restrict__ seg,
                    const int* __restrict__ idx_g,
                    const float* __restrict__ c,
                    float* __restrict__ ws_y) {
    int lane = threadIdx.x & 31;
    int wave = threadIdx.x >> 5;
    int i = blockIdx.x * 8 + wave;

    const float4* row = (const float4*)(eZ + (size_t)i * NFEAT);
    int g = seg[i];
    const float4* cp = (const float4*)(c + idx_g[g] * NFEAT);

    float4 a0 = row[lane], a1 = row[lane + 32];
    float4 b0 = cp[lane],  b1 = cp[lane + 32];
    float s = a0.x * b0.x + a0.y * b0.y + a0.z * b0.z + a0.w * b0.w
            + a1.x * b1.x + a1.y * b1.y + a1.z * b1.z + a1.w * b1.w;
    #pragma unroll
    for (int m = 16; m >= 1; m >>= 1) s += __shfl_xor(s, m, 32);

    if (lane == 0) {
        float x = s * 0.0625f;  // 1/sqrt(256)
        float y = fmaxf(x, 0.f) + log1pf(__expf(-fabsf(x)));  // stable softplus
        ws_y[i] = y;
    }
}

// ---------------------------------------------------------------------------
// denom_k: deterministic segment sum of y per graph.
// batch_segments is sorted -> binary-search [lo,hi) then block tree-reduce.
// ---------------------------------------------------------------------------
__global__ __launch_bounds__(256)
void denom_kernel(const int* __restrict__ seg,
                  const float* __restrict__ ws_y,
                  float* __restrict__ denom) {
    __shared__ int bounds[2];
    __shared__ float red[256];
    int g = blockIdx.x;
    int t = threadIdx.x;
    if (t < 2) {
        int target = g + t;
        int lo = 0, hi = NATOMS;
        while (lo < hi) {
            int mid = (lo + hi) >> 1;
            if (seg[mid] < target) lo = mid + 1; else hi = mid;
        }
        bounds[t] = lo;
    }
    __syncthreads();
    int lo = bounds[0], hi = bounds[1];
    float s = 0.f;
    for (int i = lo + t; i < hi; i += 256) s += ws_y[i];
    red[t] = s;
    __syncthreads();
    #pragma unroll
    for (int st = 128; st >= 1; st >>= 1) {
        if (t < st) red[t] += red[t + st];
        __syncthreads();
    }
    if (t == 0) denom[g] = red[0];
}

// ---------------------------------------------------------------------------
// WMMA fragment loaders (CDNA5 V_WMMA_F32_16X16X32_F16 layouts, wave32)
// ---------------------------------------------------------------------------
#define LDA 264  // 256 + 8 halfs row padding (keeps 16B alignment, spreads banks)

// A (16x32 f16): lanes 0-15 row M=lane, K = {0..7, 16..23}; lanes 16-31 same
// rows, K = {8..15, 24..31}.  Two ds_load_b128 per fragment.
__device__ __forceinline__ v16h load_a_frag(const _Float16* Alds, int lane,
                                            int mt, int kt) {
    const _Float16* p = Alds + (mt * 16 + (lane & 15)) * LDA + kt * 32
                      + ((lane >> 4) ? 8 : 0);
    v8h lo = *(const v8h*)p;
    v8h hi = *(const v8h*)(p + 16);
    v16h r;
    #pragma unroll
    for (int i = 0; i < 8; ++i) { r[i] = lo[i]; r[i + 8] = hi[i]; }
    return r;
}

// B fragments pre-swizzled in LDS: each lane's 16 halfs contiguous (32B).
__device__ __forceinline__ v16h load_b_frag(const _Float16* W, int lane,
                                            int kt, int nt) {
    return *(const v16h*)(W + ((((kt << 4) + nt) << 5) + lane) * 16);
}

__device__ __forceinline__ void gemm_stage(const _Float16* Alds,
                                           const _Float16* W, int lane,
                                           int nt0, v8f acc[4][2]) {
    #pragma unroll
    for (int kt = 0; kt < 8; ++kt) {  // K = 8 * 32 = 256
        v16h b0 = load_b_frag(W, lane, kt, nt0);
        v16h b1 = load_b_frag(W, lane, kt, nt0 + 1);
        #pragma unroll
        for (int mt = 0; mt < 4; ++mt) {
            v16h a = load_a_frag(Alds, lane, mt, kt);
            acc[mt][0] = __builtin_amdgcn_wmma_f32_16x16x32_f16(
                false, a, false, b0, (short)0, acc[mt][0], false, false);
            acc[mt][1] = __builtin_amdgcn_wmma_f32_16x16x32_f16(
                false, a, false, b1, (short)0, acc[mt][1], false, false);
        }
    }
}

// ---------------------------------------------------------------------------
// fused_mlp: per 64-atom tile:
//   a = psi*y/denom ; x = a*v ; h1 = silu(x)@W1 ; h2 = silu(h1)@W2 ; out=x+h2
// Both 256x256 f16 weight matrices resident in LDS (B-fragment swizzled),
// A tile in LDS row-major.  8 waves, each owning 2 N-tiles x 4 M-tiles.
// ---------------------------------------------------------------------------
__global__ __launch_bounds__(256)
void fused_mlp_kernel(const float* __restrict__ psi,
                      const int* __restrict__ seg,
                      const unsigned char* __restrict__ gmask,
                      const float* __restrict__ vtab_g,
                      const _Float16* __restrict__ W1h,
                      const _Float16* __restrict__ W2h,
                      const int* __restrict__ idx_g,
                      const float* __restrict__ ws_y,
                      const float* __restrict__ denom,
                      float* __restrict__ out) {
    extern __shared__ char smem[];
    _Float16* Wlds1 = (_Float16*)smem;                 // 128 KB
    _Float16* Wlds2 = Wlds1 + NFEAT * NFEAT;           // 128 KB
    _Float16* Alds  = Wlds2 + NFEAT * NFEAT;           // 64*264*2 = 33 KB
    float* vtab_s   = (float*)(Alds + 64 * LDA);       // 2 KB
    float* a_s      = vtab_s + 2 * NFEAT;              // 256 B
    int*   vidx     = (int*)(a_s + 64);                // 256 B

    const int t = threadIdx.x;
    const int atom0 = blockIdx.x * 64;

    // --- prologue: per-atom scalar a, v-table, swizzled weight staging ---
    vtab_s[t]         = vtab_g[t];
    vtab_s[t + NFEAT] = vtab_g[t + NFEAT];
    if (t < 64) {
        int i = atom0 + t;
        int g = seg[i];
        float d = gmask[g] ? denom[g] : 1.0f;
        a_s[t]  = psi[g] * ws_y[i] / d;
        vidx[t] = idx_g[g];
    }
    #pragma unroll 4
    for (int it = 0; it < NFEAT; ++it) {            // 65536 elems / 256 thr
        int idx = it * 256 + t;
        int k = idx >> 8, n = idx & 255;
        int kt = k >> 5, kl = k & 31;
        int hs = (kl >> 3) & 1, e = (kl & 7) | ((kl >> 4) << 3);
        int ln = hs * 16 + (n & 15), nt = n >> 4;
        int off = (((kt << 4) + nt) << 5 | ln) * 16 + e;
        Wlds1[off] = W1h[idx];
        Wlds2[off] = W2h[idx];
    }
    __syncthreads();

    // --- build A1 = silu(a[m] * v[idx[m]][k]) as f16 ---
    #pragma unroll 4
    for (int it = 0; it < 64; ++it) {
        int idx = it * 256 + t;
        int m = idx >> 8, kk = idx & 255;
        float xv = a_s[m] * vtab_s[vidx[m] * NFEAT + kk];
        Alds[m * LDA + kk] = (_Float16)silu_f(xv);
    }
    __syncthreads();

    const int lane = t & 31;
    const int wave = t >> 5;
    const int nt0 = wave * 2;
    const v8f vzero = {0.f, 0.f, 0.f, 0.f, 0.f, 0.f, 0.f, 0.f};

    // --- stage 1: h1 = A1 @ W1 ---
    v8f acc1[4][2];
    #pragma unroll
    for (int mt = 0; mt < 4; ++mt) { acc1[mt][0] = vzero; acc1[mt][1] = vzero; }
    gemm_stage(Alds, Wlds1, lane, nt0, acc1);
    __syncthreads();  // everyone done reading A1

    // write silu(h1) back into Alds (C/D layout: vgpr r -> M = r + 8*(lane>=16))
    #pragma unroll
    for (int mt = 0; mt < 4; ++mt)
        #pragma unroll
        for (int nt = 0; nt < 2; ++nt) {
            int n = (nt0 + nt) * 16 + (lane & 15);
            #pragma unroll
            for (int r = 0; r < 8; ++r) {
                int m = mt * 16 + r + ((lane >> 4) << 3);
                Alds[m * LDA + n] = (_Float16)silu_f(acc1[mt][nt][r]);
            }
        }
    __syncthreads();

    // --- stage 2: h2 = silu(h1) @ W2 ---
    v8f acc2[4][2];
    #pragma unroll
    for (int mt = 0; mt < 4; ++mt) { acc2[mt][0] = vzero; acc2[mt][1] = vzero; }
    gemm_stage(Alds, Wlds2, lane, nt0, acc2);

    // --- epilogue: out = x + h2, x recomputed from (a, v) ---
    #pragma unroll
    for (int mt = 0; mt < 4; ++mt)
        #pragma unroll
        for (int nt = 0; nt < 2; ++nt) {
            int n = (nt0 + nt) * 16 + (lane & 15);
            #pragma unroll
            for (int r = 0; r < 8; ++r) {
                int m = mt * 16 + r + ((lane >> 4) << 3);
                float xv = a_s[m] * vtab_s[vidx[m] * NFEAT + n];
                out[(size_t)(atom0 + m) * NFEAT + n] = xv + acc2[mt][nt][r];
            }
        }
}

// ---------------------------------------------------------------------------
// launcher
// ---------------------------------------------------------------------------
extern "C" void kernel_launch(void* const* d_in, const int* in_sizes, int n_in,
                              void* d_out, int out_size, void* d_ws, size_t ws_size,
                              hipStream_t stream) {
    // inputs (setup_inputs order); atomic_numbers (d_in[0]) is unused by ref
    const float* psi   = (const float*)d_in[1];
    const int*   seg   = (const int*)d_in[2];
    const unsigned char* gmask = (const unsigned char*)d_in[3];
    const float* eZ    = (const float*)d_in[4];
    const float* Wq    = (const float*)d_in[5];
    const float* ktab  = (const float*)d_in[6];
    const float* vtabg = (const float*)d_in[7];
    const float* W1    = (const float*)d_in[8];
    const float* W2    = (const float*)d_in[9];
    float* out = (float*)d_out;

    // workspace carve
    float* ws_y  = (float*)d_ws;                       // N floats
    float* denom = ws_y + NATOMS;                      // G floats
    int*   idx_g = (int*)(denom + NGRAPH);             // G ints
    float* cvec  = (float*)(idx_g + NGRAPH);           // 2*F floats
    _Float16* W1h = (_Float16*)(cvec + 2 * NFEAT);     // F*F halfs
    _Float16* W2h = W1h + NFEAT * NFEAT;               // F*F halfs

    prep_c_kernel<<<1, 256, 0, stream>>>(Wq, ktab, cvec);
    prep_g_kernel<<<NGRAPH / 256, 256, 0, stream>>>(psi, idx_g);
    conv_w_kernel<<<NFEAT * NFEAT / 256, 256, 0, stream>>>(W1, W2, W1h, W2h);
    score_y_kernel<<<NATOMS / 8, 256, 0, stream>>>(eZ, seg, idx_g, cvec, ws_y);
    denom_kernel<<<NGRAPH, 256, 0, stream>>>(seg, ws_y, denom);

    size_t smem = (size_t)(2 * NFEAT * NFEAT) * sizeof(_Float16)   // W1+W2
                + (size_t)(64 * LDA) * sizeof(_Float16)            // A tile
                + (size_t)(2 * NFEAT) * sizeof(float)              // v table
                + 64 * sizeof(float) + 64 * sizeof(int);           // a, idx
    fused_mlp_kernel<<<NATOMS / 64, 256, smem, stream>>>(
        psi, seg, gmask, vtabg, W1h, W2h, idx_g, ws_y, denom, out);
}
